// MemoryAugmentedNetwork_90168543412564
// MI455X (gfx1250) — matrix operand order repaired and checked by reading, not compiled, not run on hardware
//
#include <hip/hip_runtime.h>
#include <hip/hip_bf16.h>

#define BB 256
#define HH 1024
#define DD 512
#define VV 50257
#define MM 100000
#define KS 128   // k-slab staged in LDS per GEMM iteration

typedef __attribute__((ext_vector_type(16))) __bf16 bf16x16;
typedef __attribute__((ext_vector_type(8)))  float  floatx8;
typedef __attribute__((ext_vector_type(4)))  float  f32x4;
typedef __attribute__((ext_vector_type(2)))  float  f32x2;
typedef __attribute__((ext_vector_type(4)))  unsigned int u32x4;

union FragU {
    unsigned int u[8];
    u32x4        q[2];
    bf16x16      v;
};

__device__ __forceinline__ unsigned int pack_bf16(float lo, float hi) {
    unsigned int a = __float_as_uint(lo);
    unsigned int b = __float_as_uint(hi);
    a = (a + 0x7FFFu + ((a >> 16) & 1u)) >> 16;
    b = (b + 0x7FFFu + ((b >> 16) & 1u)) >> 16;
    return a | (b << 16);
}
__device__ __forceinline__ unsigned short f2bf(float f) {
    unsigned int a = __float_as_uint(f);
    return (unsigned short)((a + 0x7FFFu + ((a >> 16) & 1u)) >> 16);
}

__device__ __forceinline__ void top8_insert(float (&tv)[8], int (&ti)[8], float v, int idx) {
    if (v <= tv[7]) return;
    tv[7] = v; ti[7] = idx;
#pragma unroll
    for (int s = 7; s > 0; --s) {
        if (tv[s] > tv[s - 1]) {
            float fv = tv[s]; tv[s] = tv[s - 1]; tv[s - 1] = fv;
            int   iv = ti[s]; ti[s] = ti[s - 1]; ti[s - 1] = iv;
        }
    }
}

// Async DMA: global (bf16, raw) -> LDS, 16B per lane, tracked by ASYNCcnt.
__device__ __forceinline__ void async_b128_to_lds(unsigned int lds_off, const void* gaddr) {
    asm volatile("global_load_async_to_lds_b128 %0, %1, off"
                 :: "v"(lds_off), "v"(gaddr) : "memory");
}
__device__ __forceinline__ void wait_async0() {
    asm volatile("s_wait_asynccnt 0x0" ::: "memory");
}

// ---------------------------------------------------------------- f32 -> bf16
__global__ void __launch_bounds__(256)
cvt_f32_bf16(const float* __restrict__ in, unsigned int* __restrict__ out, int npairs) {
    int i = blockIdx.x * 256 + threadIdx.x;
    if (i < npairs) {
        f32x2 f = *(const f32x2*)(in + 2 * (size_t)i);
        out[i] = pack_bf16(f[0], f[1]);
    }
}

// ------------------------------------------------------- generic bf16 GEMM-NT
// C[256][N] = act(A_bf16[256][K] @ W_f32[N][K]^T + bias);  one block = 64 cols,
// all 256 rows.  W slab staged to LDS with inline f32->bf16 convert (read-once
// streaming of W); A slab DMA'd raw to LDS with global_load_async_to_lds_b128.
// Per k-step: batch-load B frag + all 8 A frags, then burst 8 WMMAs.
__global__ void __launch_bounds__(256)
gemm_bf16_nt(const unsigned short* __restrict__ A, int Kdim,
             const float* __restrict__ W, const float* __restrict__ bias,
             int N, int relu,
             float* __restrict__ outF, unsigned short* __restrict__ outB, int ldo) {
    __shared__ __align__(16) unsigned int wlds[64][68];   // 64 cols x 128 k (bf16 pairs)
    __shared__ __align__(16) unsigned int alds[256][68];  // 256 rows x 128 k (bf16 pairs)

    const int tid  = threadIdx.x;
    const int lane = tid & 31;
    const int wave = tid >> 5;
    const int ct   = wave & 3;   // column tile within the 64-col chunk
    const int wh   = wave >> 2;  // which half of the 16 row-tiles
    const int lm   = lane & 15;
    const int lh   = lane >> 4;
    const int colbase = blockIdx.x * 64;

    floatx8 acc[8] = {};

    const int sc = tid >> 2;          // staged col (0..63)
    const int sk = (tid & 3) * 32;    // staged k sub-range

    for (int ks = 0; ks < Kdim; ks += KS) {
        __syncthreads();
        // --- async DMA of A slab: thread tid copies row tid (256B = 16x b128)
        {
            const char* arow = (const char*)(A + (size_t)tid * Kdim + ks);
            unsigned int lrow = (unsigned int)(size_t)(void*)&alds[tid][0];
#pragma unroll
            for (int i = 0; i < 16; ++i)
                async_b128_to_lds(lrow + 16u * i, arow + 16 * i);
        }
        // --- stage W slab with f32 -> bf16 conversion (overlaps the DMA)
        int wrow = colbase + sc;
        if (wrow < N) {
            const float* wp = W + (size_t)wrow * Kdim + ks + sk;
#pragma unroll
            for (int i = 0; i < 8; ++i) {
                f32x4 f = *(const f32x4*)(wp + 4 * i);
                wlds[sc][(sk >> 1) + 2 * i]     = pack_bf16(f[0], f[1]);
                wlds[sc][(sk >> 1) + 2 * i + 1] = pack_bf16(f[2], f[3]);
            }
        } else {
#pragma unroll
            for (int i = 0; i < 16; ++i) wlds[sc][(sk >> 1) + i] = 0u;
        }
        wait_async0();
        __syncthreads();
        if (ks + KS < Kdim && wrow < N)
            __builtin_prefetch(W + (size_t)wrow * Kdim + ks + KS + sk, 0, 0);

#pragma unroll
        for (int kk = 0; kk < KS; kk += 32) {
            FragU bf;
            const u32x4* bp = (const u32x4*)&wlds[ct * 16 + lm][(kk >> 1) + lh * 4];
            bf.q[0] = bp[0];
            bf.q[1] = bp[2];
            FragU af[8];
#pragma unroll
            for (int j = 0; j < 8; ++j) {
                int m = (wh * 8 + j) * 16 + lm;
                af[j].q[0] = *(const u32x4*)&alds[m][(kk >> 1) + lh * 4];
                af[j].q[1] = *(const u32x4*)&alds[m][(kk >> 1) + 8 + lh * 4];
            }
#pragma unroll
            for (int j = 0; j < 8; ++j)
                acc[j] = __builtin_amdgcn_wmma_f32_16x16x32_bf16(
                    false, af[j].v, false, bf.v, (short)0, acc[j], false, false);
        }
    }

    int col = colbase + ct * 16 + lm;
    if (col < N) {
        float bv = bias[col];
#pragma unroll
        for (int j = 0; j < 8; ++j) {
            int rt = wh * 8 + j;
#pragma unroll
            for (int r = 0; r < 8; ++r) {
                int m = rt * 16 + r + 8 * lh;
                float v = acc[j][r] + bv;
                if (relu) v = v > 0.f ? v : 0.f;
                if (outF) outF[(size_t)m * ldo + col] = v;
                else      outB[(size_t)m * ldo + col] = f2bf(v);
            }
        }
    }
}

// ---------------------------------------------------- query row normalization
__global__ void __launch_bounds__(256)
qnorm_bf16(const float* __restrict__ Q, unsigned int* __restrict__ Qn2) {
    int wave = threadIdx.x >> 5, lane = threadIdx.x & 31;
    int row = blockIdx.x * 8 + wave;
    const float* r = Q + (size_t)row * DD + lane * 16;
    float v[16];
    float s = 0.f;
#pragma unroll
    for (int i = 0; i < 4; ++i) {
        f32x4 f = *(const f32x4*)(r + 4 * i);
#pragma unroll
        for (int c = 0; c < 4; ++c) { v[4 * i + c] = f[c]; s += f[c] * f[c]; }
    }
#pragma unroll
    for (int m = 16; m >= 1; m >>= 1) s += __shfl_xor(s, m, 32);
    float sc = 1.0f / (sqrtf(s) + 1e-8f);
#pragma unroll
    for (int j = 0; j < 8; ++j)
        Qn2[(size_t)row * (DD / 2) + lane * 8 + j] = pack_bf16(v[2 * j] * sc, v[2 * j + 1] * sc);
}

// ----------------------------------- per-key scale = importance / (norm+1e-8)
__global__ void __launch_bounds__(256)
key_scale(const float* __restrict__ Keys, const float* __restrict__ imp,
          float* __restrict__ ksc) {
    int wave = threadIdx.x >> 5, lane = threadIdx.x & 31;
    int row = blockIdx.x * 8 + wave;
    if (row >= MM) return;
    const float* r = Keys + (size_t)row * DD;
    float s = 0.f;
#pragma unroll
    for (int i = 0; i < 4; ++i) {
        f32x4 f = *(const f32x4*)(r + 4 * (lane + 32 * i));
        s += f[0] * f[0] + f[1] * f[1] + f[2] * f[2] + f[3] * f[3];
    }
#pragma unroll
    for (int m = 16; m >= 1; m >>= 1) s += __shfl_xor(s, m, 32);
    if (lane == 0) ksc[row] = imp[row] / (sqrtf(s) + 1e-8f);
}

// ------------------- sims GEMM (bf16 WMMA) fused with per-chunk top-8 select
struct SimsShared {
    union {                                     // never live simultaneously:
        __align__(16) unsigned int a[256][68];  //  - qn slab during k-loop
        float s[256][64];                       //  - sims tile after k-loop
    };
    __align__(16) unsigned int w[64][68];       // keys slab (bf16 pairs)
};

__global__ void __launch_bounds__(256)
sims_topk(const unsigned short* __restrict__ Qn, const float* __restrict__ Keys,
          const float* __restrict__ ksc,
          float* __restrict__ pvals, int* __restrict__ pidx) {
    __shared__ SimsShared sh;

    const int tid  = threadIdx.x;
    const int lane = tid & 31;
    const int wave = tid >> 5;
    const int ct   = wave & 3;
    const int wh   = wave >> 2;
    const int lm   = lane & 15;
    const int lh   = lane >> 4;
    const int colbase = blockIdx.x * 64;

    floatx8 acc[8] = {};
    const int sc = tid >> 2;
    const int sk = (tid & 3) * 32;

    for (int ks = 0; ks < DD; ks += KS) {
        __syncthreads();
        // --- async DMA of qn slab (bf16, raw)
        {
            const char* arow = (const char*)(Qn + (size_t)tid * DD + ks);
            unsigned int lrow = (unsigned int)(size_t)(void*)&sh.a[tid][0];
#pragma unroll
            for (int i = 0; i < 16; ++i)
                async_b128_to_lds(lrow + 16u * i, arow + 16 * i);
        }
        // --- stage keys slab with f32 -> bf16 conversion
        int wrow = colbase + sc;
        if (wrow < MM) {
            const float* wp = Keys + (size_t)wrow * DD + ks + sk;
#pragma unroll
            for (int i = 0; i < 8; ++i) {
                f32x4 f = *(const f32x4*)(wp + 4 * i);
                sh.w[sc][(sk >> 1) + 2 * i]     = pack_bf16(f[0], f[1]);
                sh.w[sc][(sk >> 1) + 2 * i + 1] = pack_bf16(f[2], f[3]);
            }
        } else {
#pragma unroll
            for (int i = 0; i < 16; ++i) sh.w[sc][(sk >> 1) + i] = 0u;
        }
        wait_async0();
        __syncthreads();
        if (ks + KS < DD && wrow < MM)
            __builtin_prefetch(Keys + (size_t)wrow * DD + ks + KS + sk, 0, 0);

#pragma unroll
        for (int kk = 0; kk < KS; kk += 32) {
            FragU bf;
            const u32x4* bp = (const u32x4*)&sh.w[ct * 16 + lm][(kk >> 1) + lh * 4];
            bf.q[0] = bp[0];
            bf.q[1] = bp[2];
            FragU af[8];
#pragma unroll
            for (int j = 0; j < 8; ++j) {
                int m = (wh * 8 + j) * 16 + lm;
                af[j].q[0] = *(const u32x4*)&sh.a[m][(kk >> 1) + lh * 4];
                af[j].q[1] = *(const u32x4*)&sh.a[m][(kk >> 1) + 8 + lh * 4];
            }
#pragma unroll
            for (int j = 0; j < 8; ++j)
                acc[j] = __builtin_amdgcn_wmma_f32_16x16x32_bf16(
                    false, af[j].v, false, bf.v, (short)0, acc[j], false, false);
        }
    }

    __syncthreads();   // retire slabs before aliasing LDS as sims tile
    int c = ct * 16 + lm;
    int colg = colbase + c;
    float scale = (colg < MM) ? ksc[colg] : 0.f;
#pragma unroll
    for (int j = 0; j < 8; ++j) {
        int rt = wh * 8 + j;
#pragma unroll
        for (int r = 0; r < 8; ++r) {
            int m = rt * 16 + r + 8 * lh;
            sh.s[m][c] = acc[j][r] * scale;
        }
    }
    __syncthreads();

    // per-query top-8 over this 64-key chunk
    float tv[8]; int ti[8];
#pragma unroll
    for (int j = 0; j < 8; ++j) { tv[j] = -__builtin_inff(); ti[j] = 0; }
    int q = tid;
    for (int cc = 0; cc < 64; ++cc) {
        int cl = (cc + q) & 63;       // staggered to spread LDS banks
        int cg = colbase + cl;
        float v = (cg < MM) ? sh.s[q][cl] : -__builtin_inff();
        top8_insert(tv, ti, v, cg);
    }
    size_t off = ((size_t)blockIdx.x * BB + q) * 8;
#pragma unroll
    for (int j = 0; j < 8; ++j) { pvals[off + j] = tv[j]; pidx[off + j] = ti[j]; }
}

// -------------------------------- merge partial top-8 lists + softmax weights
__global__ void __launch_bounds__(256)
topk_merge_softmax(const float* __restrict__ pvals, const int* __restrict__ pidx,
                   int S, float* __restrict__ attn, int* __restrict__ topidx) {
    __shared__ float lv[256][8];
    __shared__ int   li[256][8];
    int b = blockIdx.x, tid = threadIdx.x;

    float tv[8]; int ti[8];
#pragma unroll
    for (int j = 0; j < 8; ++j) { tv[j] = -__builtin_inff(); ti[j] = 0; }
    int total = S * 8;
    for (int c = tid; c < total; c += 256) {
        int s = c >> 3, j = c & 7;
        size_t off = ((size_t)s * BB + b) * 8 + j;
        top8_insert(tv, ti, pvals[off], pidx[off]);
    }
#pragma unroll
    for (int j = 0; j < 8; ++j) { lv[tid][j] = tv[j]; li[tid][j] = ti[j]; }
    __syncthreads();

    if (tid < 32) {
#pragma unroll
        for (int j = 0; j < 8; ++j) { tv[j] = -__builtin_inff(); ti[j] = 0; }
        for (int g = 0; g < 8; ++g) {
            int r = tid + g * 32;
#pragma unroll
            for (int j = 0; j < 8; ++j) top8_insert(tv, ti, lv[r][j], li[r][j]);
        }
#pragma unroll
        for (int j = 0; j < 8; ++j) { lv[tid][j] = tv[j]; li[tid][j] = ti[j]; }
    }
    __syncthreads();

    if (tid == 0) {
#pragma unroll
        for (int j = 0; j < 8; ++j) { tv[j] = -__builtin_inff(); ti[j] = 0; }
        for (int r = 0; r < 32; ++r)
#pragma unroll
            for (int j = 0; j < 8; ++j) top8_insert(tv, ti, lv[r][j], li[r][j]);
        float mx = tv[0], sum = 0.f, e[8];
#pragma unroll
        for (int j = 0; j < 8; ++j) { e[j] = __expf(tv[j] - mx); sum += e[j]; }
        float inv = 1.f / sum;
#pragma unroll
        for (int j = 0; j < 8; ++j) { attn[b * 8 + j] = e[j] * inv; topidx[b * 8 + j] = ti[j]; }
    }
}

// -------------------- combined = concat(hidden, attn-weighted gather) as bf16
__global__ void __launch_bounds__(256)
build_combined(const float* __restrict__ hidden, const float* __restrict__ mvals,
               const float* __restrict__ attn, const int* __restrict__ topidx,
               unsigned short* __restrict__ comb) {
    int b = blockIdx.x, tid = threadIdx.x;
    float a[8]; int id[8];
#pragma unroll
    for (int k = 0; k < 8; ++k) { a[k] = attn[b * 8 + k]; id[k] = topidx[b * 8 + k]; }
    for (int e = tid; e < HH + DD; e += 256) {
        float v;
        if (e < HH) {
            v = hidden[(size_t)b * HH + e];
        } else {
            int d = e - HH;
            v = 0.f;
#pragma unroll
            for (int k = 0; k < 8; ++k) v += a[k] * mvals[(size_t)id[k] * DD + d];
        }
        comb[(size_t)b * (HH + DD) + e] = f2bf(v);
    }
}

// ----------------------------------------------------------------- value head
__global__ void __launch_bounds__(256)
value_head(const float* __restrict__ hv, const float* __restrict__ Wv2,
           const float* __restrict__ bv2, float* __restrict__ out) {
    int wave = threadIdx.x >> 5, lane = threadIdx.x & 31;
    int b = blockIdx.x * 8 + wave;
    const float* r = hv + (size_t)b * HH;
    float s = 0.f;
#pragma unroll
    for (int i = 0; i < 32; ++i) s += r[lane + 32 * i] * Wv2[lane + 32 * i];
#pragma unroll
    for (int m = 16; m >= 1; m >>= 1) s += __shfl_xor(s, m, 32);
    if (lane == 0) out[b] = s + bv2[0];
}

// ============================================================================
extern "C" void kernel_launch(void* const* d_in, const int* in_sizes, int n_in,
                              void* d_out, int out_size, void* d_ws, size_t ws_size,
                              hipStream_t stream) {
    const float* hidden = (const float*)d_in[0];
    const float* keys   = (const float*)d_in[1];
    const float* mvals  = (const float*)d_in[2];
    const float* imp    = (const float*)d_in[3];
    const float* Wq     = (const float*)d_in[4];
    const float* bq     = (const float*)d_in[5];
    const float* Wp1    = (const float*)d_in[6];
    const float* bp1    = (const float*)d_in[7];
    const float* Wp2    = (const float*)d_in[8];
    const float* bp2    = (const float*)d_in[9];
    const float* Wv1    = (const float*)d_in[10];
    const float* bv1    = (const float*)d_in[11];
    const float* Wv2    = (const float*)d_in[12];
    const float* bv2    = (const float*)d_in[13];
    float* out = (float*)d_out;

    constexpr int    S_CHUNKS = (MM + 63) / 64;   // 1563
    constexpr size_t OFF_HB   = 0;                                    // hidden bf16
    constexpr size_t OFF_Q    = OFF_HB + (size_t)BB * HH * 2;         // query f32
    constexpr size_t OFF_QN   = OFF_Q + (size_t)BB * DD * 4;          // qn bf16
    constexpr size_t OFF_KSC  = OFF_QN + (size_t)BB * DD * 2;         // key scales
    constexpr size_t OFF_PV   = (OFF_KSC + (size_t)MM * 4 + 255) & ~(size_t)255;
    constexpr size_t OFF_PI   = OFF_PV + (size_t)S_CHUNKS * BB * 8 * 4;
    constexpr size_t OFF_ATTN = OFF_PI + (size_t)S_CHUNKS * BB * 8 * 4;
    constexpr size_t OFF_TIX  = OFF_ATTN + (size_t)BB * 8 * 4;
    constexpr size_t OFF_COMB = OFF_TIX + (size_t)BB * 8 * 4;         // combined bf16
    constexpr size_t OFF_HP   = OFF_COMB + (size_t)BB * (HH + DD) * 2; // h_p bf16
    constexpr size_t OFF_HV   = OFF_HP + (size_t)BB * HH * 2;          // h_v f32

    char* ws = (char*)d_ws;
    unsigned short* p_hb   = (unsigned short*)(ws + OFF_HB);
    float*          p_q    = (float*)(ws + OFF_Q);
    unsigned short* p_qn   = (unsigned short*)(ws + OFF_QN);
    float*          p_ksc  = (float*)(ws + OFF_KSC);
    float*          p_pv   = (float*)(ws + OFF_PV);
    int*            p_pi   = (int*)(ws + OFF_PI);
    float*          p_attn = (float*)(ws + OFF_ATTN);
    int*            p_tix  = (int*)(ws + OFF_TIX);
    unsigned short* p_comb = (unsigned short*)(ws + OFF_COMB);
    unsigned short* p_hp   = (unsigned short*)(ws + OFF_HP);
    float*          p_hv   = (float*)(ws + OFF_HV);

    // 1. hidden -> bf16
    cvt_f32_bf16<<<(BB * HH / 2 + 255) / 256, 256, 0, stream>>>(
        hidden, (unsigned int*)p_hb, BB * HH / 2);
    // 2. query = hidden @ Wq^T + bq
    gemm_bf16_nt<<<DD / 64, 256, 0, stream>>>(p_hb, HH, Wq, bq, DD, 0, p_q, nullptr, DD);
    // 3. qn (normalized query, bf16)
    qnorm_bf16<<<BB / 8, 256, 0, stream>>>(p_q, (unsigned int*)p_qn);
    // 4. per-key scale = importance / (||key|| + 1e-8)
    key_scale<<<(MM + 7) / 8, 256, 0, stream>>>(keys, imp, p_ksc);
    // 5. sims GEMM fused with per-chunk top-8 (keys streamed once)
    sims_topk<<<S_CHUNKS, 256, 0, stream>>>(p_qn, keys, p_ksc, p_pv, p_pi);
    // 6. global top-8 merge + softmax
    topk_merge_softmax<<<BB, 256, 0, stream>>>(p_pv, p_pi, S_CHUNKS, p_attn, p_tix);
    // 7. combined = concat(hidden, context) as bf16
    build_combined<<<BB, 256, 0, stream>>>(hidden, mvals, p_attn, p_tix, p_comb);
    // 8. h_p = relu(combined @ Wp1^T + bp1) -> bf16
    gemm_bf16_nt<<<HH / 64, 256, 0, stream>>>(p_comb, HH + DD, Wp1, bp1, HH, 1,
                                              nullptr, p_hp, HH);
    // 9. h_v = relu(combined @ Wv1^T + bv1) -> f32
    gemm_bf16_nt<<<HH / 64, 256, 0, stream>>>(p_comb, HH + DD, Wv1, bv1, HH, 1,
                                              p_hv, nullptr, HH);
    // 10. logits = h_p @ Wp2^T + bp2 -> d_out (Wp2 streamed once)
    gemm_bf16_nt<<<(VV + 63) / 64, 256, 0, stream>>>(p_hp, HH, Wp2, bp2, VV, 0,
                                                     out, nullptr, VV);
    // 11. value head -> d_out[B*V ..]
    value_head<<<BB / 8, 256, 0, stream>>>(p_hv, Wv2, bv2, out + (size_t)BB * VV);
}
